// MultiHeadGQAttention_53377853554773
// MI455X (gfx1250) — matrix operand order repaired and checked by reading, not compiled, not run on hardware
//
#include <hip/hip_runtime.h>
#include <hip/hip_bf16.h>

// ---------------------------------------------------------------------------
// CDNA5 (gfx1250) GQA attention forward, wave32 + v_wmma_f32_16x16x32_f16,
// async global->LDS staging (ASYNCcnt) where no transpose is required.
// ---------------------------------------------------------------------------

typedef __attribute__((ext_vector_type(16))) _Float16 v16h;
typedef __attribute__((ext_vector_type(8)))  float    v8f;

union Frag16 {            // 8 VGPRs = 16 halves
    v16h     v;
    _Float16 h[16];
    unsigned u32[8];
    uint4    q[2];
};

static __device__ __forceinline__ v8f wmma_f16(const v16h& a, const v16h& b, const v8f& c) {
    return __builtin_amdgcn_wmma_f32_16x16x32_f16(false, a, false, b, (short)0, c, false, false);
}

// A-fragment (16x32 f16, M x K): lane L holds row m = L%16.
// VGPR v (halves 2v,2v+1) holds k = 2v + 16*(L/16) (+0/+1).
static __device__ __forceinline__ int a_frag_kbase(int v, int g) {
    return 2 * v + 16 * g;
}

// --- async global->LDS path (GLOBAL_LOAD_ASYNC_TO_LDS_B128 / ASYNCcnt) -----
#if defined(__gfx1250__) && __has_builtin(__builtin_amdgcn_global_load_async_to_lds_b128)
#define HAVE_ASYNC_LDS 1
#else
#define HAVE_ASYNC_LDS 0
#endif

// Builtin signature (from hipcc diagnostic): param0 = AS1 pointer to
// gcc-vector 'int __vector(4)', param1 = AS3 pointer to same, then two
// immediate ints (offset, cpol). Use the __vector_size__ spelling so the
// pointee type matches the builtin's generic-vector type exactly.
typedef int v4i_vs __attribute__((__vector_size__(16)));
typedef __attribute__((address_space(1))) v4i_vs* glb_v4i_p;
typedef __attribute__((address_space(3))) v4i_vs* lds_v4i_p;

static __device__ __forceinline__ glb_v4i_p glb_cast16(const void* p) {
    return (glb_v4i_p)(unsigned long long)p;
}
static __device__ __forceinline__ lds_v4i_p lds_cast16(void* p) {
    return (lds_v4i_p)(unsigned)(unsigned long long)p;
}
static __device__ __forceinline__ void wait_async0() {
#if __has_builtin(__builtin_amdgcn_s_wait_asynccnt)
    __builtin_amdgcn_s_wait_asynccnt(0);
#elif defined(__gfx1250__)
    asm volatile("s_wait_asynccnt 0x0" ::: "memory");
#endif
}

// copy 16 bytes global -> LDS (async DMA if available, else via VGPRs)
static __device__ __forceinline__ void copy16_g2l(void* lds_dst, const void* gsrc) {
#if HAVE_ASYNC_LDS
    __builtin_amdgcn_global_load_async_to_lds_b128(glb_cast16(gsrc), lds_cast16(lds_dst), 0, 0);
#else
    *(uint4*)lds_dst = *(const uint4*)gsrc;
#endif
}
static __device__ __forceinline__ void copy16_fence() {
#if HAVE_ASYNC_LDS
    wait_async0();
#endif
}

// ---------------------------------------------------------------------------
// f32 -> f16 conversion
// ---------------------------------------------------------------------------
__global__ void f32_to_f16_kernel(const float* __restrict__ in,
                                  _Float16* __restrict__ out, size_t n) {
    size_t i = (size_t)blockIdx.x * blockDim.x + threadIdx.x;
    if (i < n) out[i] = (_Float16)in[i];
}

// ---------------------------------------------------------------------------
// Tiled WMMA GEMM: C[MxN] = A[MxK] @ B[KxN], A/B f16 row-major.
// Block = 128 threads (4 waves). Tile 64x64, BK=32. Wave w does rows w*16..+16.
// A tile staged via async global->LDS DMA; B tile transposed through VGPRs
// (overlapped with the in-flight async copy).
// ---------------------------------------------------------------------------
template <bool OUT_HALF>
__global__ __launch_bounds__(128) void gemm_f16_kernel(
    const _Float16* __restrict__ A, const _Float16* __restrict__ B,
    void* __restrict__ Cout, int M, int N, int K) {
    constexpr int BM = 64, BN = 64, BK = 32;
    __shared__ _Float16 sA[BM * BK];   // [m][k] row-major
    __shared__ _Float16 sBT[BN * BK];  // [n][k] (B transposed)

    const int tid  = threadIdx.x;
    const int wave = tid >> 5;
    const int lane = tid & 31;
    const int g    = lane >> 4;
    const int ml   = lane & 15;
    const int m0   = blockIdx.y * BM;
    const int n0   = blockIdx.x * BN;

    v8f acc[4] = {};

    for (int k0 = 0; k0 < K; k0 += BK) {
        // --- issue async A tile copy first: 64x32 halves, 2 x b128/thread ---
#pragma unroll
        for (int i = 0; i < 2; ++i) {
            int idx8 = tid * 2 + i;            // 256 groups of 8 halves
            int r    = idx8 >> 2;              // 4 groups per row
            int c8   = (idx8 & 3) * 8;
            const _Float16* src = A + (size_t)(m0 + r) * K + k0 + c8;
            copy16_g2l(&sA[r * BK + c8], src);
        }
        // --- B tile transposed through VGPRs (overlaps async A copy) ---
#pragma unroll
        for (int i = 0; i < 2; ++i) {
            int idx8 = tid * 2 + i;
            int kk   = idx8 >> 3;              // 8 groups per K-row
            int nn   = (idx8 & 7) * 8;
            const _Float16* src = B + (size_t)(k0 + kk) * N + n0 + nn;
            uint4 d = *(const uint4*)src;
            const _Float16* hp = (const _Float16*)&d;
#pragma unroll
            for (int j = 0; j < 8; ++j) sBT[(nn + j) * BK + kk] = hp[j];
            if (k0 + BK < K) __builtin_prefetch(src + (size_t)BK * N, 0, 1);
        }
        copy16_fence();
        __syncthreads();

        // --- A fragment for this wave's 16 rows ---
        Frag16 af;
        const int am = wave * 16 + ml;
#pragma unroll
        for (int v = 0; v < 8; ++v)
            af.u32[v] = *(const unsigned*)&sA[am * BK + a_frag_kbase(v, g)];

        // --- 4 N-subtiles ---
#pragma unroll
        for (int nt = 0; nt < 4; ++nt) {
            Frag16 bf;
            const int bn = nt * 16 + ml;
            const int ks = 16 * g;
            bf.q[0] = *(const uint4*)&sBT[bn * BK + ks];
            bf.q[1] = *(const uint4*)&sBT[bn * BK + ks + 8];
            acc[nt] = wmma_f16(af.v, bf.v, acc[nt]);
        }
        __syncthreads();
    }

    // --- epilogue: C layout VGPR r -> row r + 8*g, col lane%16 ---
#pragma unroll
    for (int nt = 0; nt < 4; ++nt) {
#pragma unroll
        for (int r = 0; r < 8; ++r) {
            int row   = m0 + wave * 16 + r + 8 * g;
            int col   = n0 + nt * 16 + ml;
            float val = acc[nt][r];
            if (OUT_HALF)
                ((_Float16*)Cout)[(size_t)row * N + col] = (_Float16)val;
            else
                ((float*)Cout)[(size_t)row * N + col] = val;
        }
    }
}

// ---------------------------------------------------------------------------
// RoPE epilogue: (B,S,Hn,hd) f32 -> roped f16. One thread per complex pair.
// ---------------------------------------------------------------------------
__global__ void rope_f32_to_f16_kernel(const float* __restrict__ xin,
                                       const float* __restrict__ fcos,
                                       const float* __restrict__ fsin,
                                       _Float16* __restrict__ xout,
                                       int S, int Hn, int hp, size_t npairs) {
    size_t i = (size_t)blockIdx.x * blockDim.x + threadIdx.x;
    if (i >= npairs) return;
    int    j   = (int)(i % hp);
    size_t rem = i / hp;              // (b*S + s)*Hn + h
    int    s   = (int)((rem / Hn) % S);
    float  c   = fcos[(size_t)s * hp + j];
    float  sn  = fsin[(size_t)s * hp + j];
    size_t base = rem * (size_t)(2 * hp) + 2 * j;
    float re = xin[base], im = xin[base + 1];
    xout[base]     = (_Float16)(re * c - im * sn);
    xout[base + 1] = (_Float16)(re * sn + im * c);
}

// ---------------------------------------------------------------------------
// Flash attention (causal, GQA). Grid: (B*H, S/64). Block: 128 threads.
// Wave w owns query rows qblk*64 + w*16 .. +16; KV iterated in 32-wide blocks.
// K tile staged by async DMA (consumed row-major == B^T for Q.K^T);
// V transposed through VGPRs while the DMA is in flight.
// ---------------------------------------------------------------------------
__global__ __launch_bounds__(128) void flash_attn_kernel(
    const _Float16* __restrict__ Q, const _Float16* __restrict__ Km,
    const _Float16* __restrict__ Vm, _Float16* __restrict__ O,
    int S, int Hq, int Hkv) {
    constexpr int HD = 128;
    const float scale = 0.08838834764831845f;  // 1/sqrt(128)

    const int bh   = blockIdx.x;
    const int h    = bh % Hq;
    const int b    = bh / Hq;
    const int hk   = h / (Hq / Hkv);
    const int qblk = blockIdx.y;
    const int wave = threadIdx.x >> 5;
    const int lane = threadIdx.x & 31;
    const int g    = lane >> 4;
    const int ml   = lane & 15;
    const int qrow0 = qblk * 64 + wave * 16;

    __shared__ _Float16 sK[32 * HD];       // [kv][hd]  (== B^T for Q.K^T)
    __shared__ _Float16 sVT[HD * 32];      // [hd][kv]  (== B^T for P.V)
    __shared__ _Float16 sP[4][16 * 32];    // per-wave P tile (C->A relayout)

    const size_t qstride = (size_t)Hq * HD;
    const size_t kstride = (size_t)Hkv * HD;
    const _Float16* qbase = Q  + ((size_t)b * S * Hq  + h)  * HD;
    const _Float16* kbase = Km + ((size_t)b * S * Hkv + hk) * HD;
    const _Float16* vbase = Vm + ((size_t)b * S * Hkv + hk) * HD;

    // Resident Q fragments: 4 hd-chunks of 32
    Frag16 qf[4];
    {
        const _Float16* qr = qbase + (size_t)(qrow0 + ml) * qstride;
#pragma unroll
        for (int c = 0; c < 4; ++c)
#pragma unroll
            for (int v = 0; v < 8; ++v)
                qf[c].u32[v] = *(const unsigned*)(qr + c * 32 + a_frag_kbase(v, g));
    }

    v8f   o[8] = {};
    float mrow[8], lrow[8];
#pragma unroll
    for (int r = 0; r < 8; ++r) { mrow[r] = -1e30f; lrow[r] = 0.0f; }

    const int kvend = qblk * 64 + 64;  // causal limit for this 64-row block
    for (int kv0 = 0; kv0 < kvend; kv0 += 32) {
        __syncthreads();
        // --- issue async K tile (row-major, 32x128 halves, 4 x b128/thread) ---
#pragma unroll
        for (int i = 0; i < 4; ++i) {
            int idx8 = threadIdx.x * 4 + i;   // 512 groups of 8 halves
            int r    = idx8 >> 4;             // 16 groups per row
            int c8   = (idx8 & 15) * 8;
            copy16_g2l(&sK[r * HD + c8],
                       kbase + (size_t)(kv0 + r) * kstride + c8);
        }
        // --- V transposed through VGPRs (overlaps async K copy) ---
#pragma unroll
        for (int i = 0; i < 4; ++i) {
            int idx8 = threadIdx.x * 4 + i;
            int r    = idx8 >> 4;
            int c8   = (idx8 & 15) * 8;
            uint4 d = *(const uint4*)(vbase + (size_t)(kv0 + r) * kstride + c8);
            const _Float16* hp = (const _Float16*)&d;
#pragma unroll
            for (int j = 0; j < 8; ++j) sVT[(c8 + j) * 32 + r] = hp[j];
        }
        copy16_fence();
        __syncthreads();

        // --- scores: 16x32 tile = 2 C-fragments, K-loop over hd (4 x 32) ---
        v8f sc[2] = {};
#pragma unroll
        for (int nt = 0; nt < 2; ++nt) {
#pragma unroll
            for (int c = 0; c < 4; ++c) {
                Frag16 bf;
                const int bn = nt * 16 + ml;
                const int ks = c * 32 + 16 * g;
                bf.q[0] = *(const uint4*)&sK[bn * HD + ks];
                bf.q[1] = *(const uint4*)&sK[bn * HD + ks + 8];
                sc[nt] = wmma_f16(qf[c].v, bf.v, sc[nt]);
            }
        }

        // --- online softmax (row = qrow0 + r + 8g, cols kv0+ml, kv0+16+ml) ---
#pragma unroll
        for (int r = 0; r < 8; ++r) {
            const int row = qrow0 + r + 8 * g;
            float v0 = sc[0][r] * scale;
            float v1 = sc[1][r] * scale;
            if (kv0 + ml > row)      v0 = -1e30f;
            if (kv0 + 16 + ml > row) v1 = -1e30f;

            float mx = fmaxf(v0, v1);
#pragma unroll
            for (int mk = 1; mk <= 8; mk <<= 1) mx = fmaxf(mx, __shfl_xor(mx, mk, 32));
            const float mnew  = fmaxf(mrow[r], mx);
            const float alpha = __expf(mrow[r] - mnew);
            const float p0 = __expf(v0 - mnew);
            const float p1 = __expf(v1 - mnew);

            sP[wave][(r + 8 * g) * 32 + ml]      = (_Float16)p0;
            sP[wave][(r + 8 * g) * 32 + 16 + ml] = (_Float16)p1;

            float ps = p0 + p1;
#pragma unroll
            for (int mk = 1; mk <= 8; mk <<= 1) ps += __shfl_xor(ps, mk, 32);
            lrow[r] = lrow[r] * alpha + ps;
            mrow[r] = mnew;
#pragma unroll
            for (int t = 0; t < 8; ++t) o[t][r] = o[t][r] * alpha;
        }

        // --- P(16x32) @ V(32x128): reload P as A-fragment from own LDS tile ---
        Frag16 pa;
#pragma unroll
        for (int v = 0; v < 8; ++v)
            pa.u32[v] = *(const unsigned*)&sP[wave][ml * 32 + a_frag_kbase(v, g)];
#pragma unroll
        for (int t = 0; t < 8; ++t) {
            Frag16 bf;
            const int bn = t * 16 + ml;
            bf.q[0] = *(const uint4*)&sVT[bn * 32 + 16 * g];
            bf.q[1] = *(const uint4*)&sVT[bn * 32 + 16 * g + 8];
            o[t] = wmma_f16(pa.v, bf.v, o[t]);
        }
    }

    // --- normalize and store (B,S,H,hd) f16 ---
    _Float16* obase = O + ((size_t)b * S * Hq + h) * HD;
#pragma unroll
    for (int r = 0; r < 8; ++r) {
        const float inv = 1.0f / lrow[r];
        const int   row = qrow0 + r + 8 * g;
#pragma unroll
        for (int t = 0; t < 8; ++t)
            obase[(size_t)row * qstride + t * 16 + ml] = (_Float16)(o[t][r] * inv);
    }
}

// ---------------------------------------------------------------------------
// Host-side launcher
// ---------------------------------------------------------------------------
static inline void* bump(char*& p, size_t bytes) {
    void* r = p;
    p += (bytes + 255) & ~(size_t)255;
    return r;
}

extern "C" void kernel_launch(void* const* d_in, const int* in_sizes, int n_in,
                              void* d_out, int out_size, void* d_ws, size_t ws_size,
                              hipStream_t stream) {
    (void)in_sizes; (void)n_in; (void)out_size; (void)ws_size;

    const float* q    = (const float*)d_in[0];
    const float* k    = (const float*)d_in[1];
    const float* v    = (const float*)d_in[2];
    // d_in[3] = mask (causal; implicit in kernel)
    const float* fcos = (const float*)d_in[4];
    const float* fsin = (const float*)d_in[5];
    const float* Wq   = (const float*)d_in[6];
    const float* Wk   = (const float*)d_in[7];
    const float* Wv   = (const float*)d_in[8];
    const float* Wo   = (const float*)d_in[9];

    constexpr int B = 2, S = 2048, D = 2048, H = 16, G = 4;
    constexpr int Hkv = H / G, hd = D / H, Dkv = D / G;
    constexpr size_t M = (size_t)B * S;  // 4096 rows

    char* p = (char*)d_ws;
    _Float16* q16   = (_Float16*)bump(p, M * D * 2);
    _Float16* k16   = (_Float16*)bump(p, M * D * 2);
    _Float16* v16   = (_Float16*)bump(p, M * D * 2);
    _Float16* Wq16  = (_Float16*)bump(p, (size_t)D * D * 2);
    _Float16* Wk16  = (_Float16*)bump(p, (size_t)D * Dkv * 2);
    _Float16* Wv16  = (_Float16*)bump(p, (size_t)D * Dkv * 2);
    _Float16* Wo16  = (_Float16*)bump(p, (size_t)D * D * 2);
    float*    qp32  = (float*)bump(p, M * D * 4);
    float*    kp32  = (float*)bump(p, M * Dkv * 4);
    _Float16* qr16  = (_Float16*)bump(p, M * D * 2);
    _Float16* kr16  = (_Float16*)bump(p, M * Dkv * 2);
    _Float16* vp16  = (_Float16*)bump(p, M * Dkv * 2);
    _Float16* at16  = (_Float16*)bump(p, M * D * 2);

    auto cvt = [&](const float* src, _Float16* dst, size_t n) {
        f32_to_f16_kernel<<<dim3((unsigned)((n + 255) / 256)), 256, 0, stream>>>(src, dst, n);
    };
    cvt(q, q16, M * D);
    cvt(k, k16, M * D);
    cvt(v, v16, M * D);
    cvt(Wq, Wq16, (size_t)D * D);
    cvt(Wk, Wk16, (size_t)D * Dkv);
    cvt(Wv, Wv16, (size_t)D * Dkv);
    cvt(Wo, Wo16, (size_t)D * D);

    // Projections
    gemm_f16_kernel<false><<<dim3(D / 64, (unsigned)(M / 64)), 128, 0, stream>>>(
        q16, Wq16, qp32, (int)M, D, D);
    gemm_f16_kernel<false><<<dim3(Dkv / 64, (unsigned)(M / 64)), 128, 0, stream>>>(
        k16, Wk16, kp32, (int)M, Dkv, D);
    gemm_f16_kernel<true><<<dim3(Dkv / 64, (unsigned)(M / 64)), 128, 0, stream>>>(
        v16, Wv16, vp16, (int)M, Dkv, D);

    // RoPE
    {
        size_t npq = M * (size_t)H * (hd / 2);
        rope_f32_to_f16_kernel<<<dim3((unsigned)((npq + 255) / 256)), 256, 0, stream>>>(
            qp32, fcos, fsin, qr16, S, H, hd / 2, npq);
        size_t npk = M * (size_t)Hkv * (hd / 2);
        rope_f32_to_f16_kernel<<<dim3((unsigned)((npk + 255) / 256)), 256, 0, stream>>>(
            kp32, fcos, fsin, kr16, S, Hkv, hd / 2, npk);
    }

    // Causal GQA flash attention
    flash_attn_kernel<<<dim3(B * H, S / 64), 128, 0, stream>>>(
        qr16, kr16, vp16, at16, S, H, Hkv);

    // Output projection -> d_out (f32)
    gemm_f16_kernel<false><<<dim3(D / 64, (unsigned)(M / 64)), 128, 0, stream>>>(
        at16, Wo16, (float*)d_out, (int)M, D, D);
}
